// Attention_68247030333530
// MI455X (gfx1250) — matrix operand order repaired and checked by reading, not compiled, not run on hardware
//
#include <hip/hip_runtime.h>
#include <hip/hip_bf16.h>

#define B_ 256
#define N_ 196
#define E_ 2048
#define H_ 512

typedef __attribute__((ext_vector_type(16))) __bf16 v16bf;
typedef __attribute__((ext_vector_type(8)))  float  v8f;

// fp32 pair -> packed bf16 dword. Round-to-nearest via +0x8000, then a single
// v_perm_b32 selects the two high halves (bytes {a2,a3,b2,b3}).
__device__ __forceinline__ unsigned int pack_bf16(float a, float b) {
    unsigned int ua = __float_as_uint(a) + 0x8000u;
    unsigned int ub = __float_as_uint(b) + 0x8000u;
    return __builtin_amdgcn_perm(ub, ua, 0x07060302u);
}

// ---------------------------------------------------------------------------
// Phase 0: repack W_w (E x H fp32, row-major) into bf16 WMMA B-fragment layout.
// Tile (kb: 64 K-blocks of 32, ct: 32 col-tiles of 16). Per tile, 32 lanes x 8
// dwords. Lane l<16 holds col=ct*16+l, K = kb*32 + {0..15}; lane>=16 holds the
// same col with K = kb*32 + {16..31} (contiguous k order, same as the A fragment).
// ---------------------------------------------------------------------------
__global__ __launch_bounds__(256) void pack_w_kernel(
        const float* __restrict__ Ww, unsigned int* __restrict__ Wbf) {
    int i  = blockIdx.x * 256 + threadIdx.x;   // 0 .. 524287 dwords
    int d  = i & 7;
    int l  = (i >> 3) & 31;
    int ct = (i >> 8) & 31;
    int kb = i >> 13;
    int c  = ct * 16 + (l & 15);
    int k  = kb * 32 + ((l >> 4) ? 16 : 0) + 2 * d;
    Wbf[i] = pack_bf16(Ww[k * H_ + c], Ww[(k + 1) * H_ + c]);
}

// ---------------------------------------------------------------------------
// Phase 1: uh[b][h] = hidden[b]·U_w[:,h] + U_b[h] + W_b[h]   (biases folded)
// ---------------------------------------------------------------------------
__global__ __launch_bounds__(256) void uh_kernel(
        const float* __restrict__ hidden, const float* __restrict__ Uw,
        const float* __restrict__ Ub, const float* __restrict__ Wb,
        float* __restrict__ uh) {
    int h = blockIdx.x * 256 + threadIdx.x;
    int b = blockIdx.y;
    float s = Ub[h] + Wb[h];
    const float* hv = hidden + b * H_;
    #pragma unroll 4
    for (int k = 0; k < H_; ++k) s = fmaf(hv[k], Uw[k * H_ + h], s);
    uh[b * H_ + h] = s;
}

// ---------------------------------------------------------------------------
// Phase 2: main fused kernel. Block = 16-row strip of flattened (b,n).
// 256 threads = 8 waves; each wave owns 64 of H=512 (4 WMMA tiles, 32 acc
// VGPRs -> in-place WMMA). A fragments are built per-lane from global (one
// 128B line per row per k-step, absorbed by WGP$). The k-loop is unrolled 2x
// so each pipeline stage gets fresh virtual registers (breaks the loop-carried
// tuple-reuse conflict that forced acc live-range splits). Barrier-free.
// ---------------------------------------------------------------------------
__global__ __launch_bounds__(256, 1) void attn_gemm_kernel(
        const float* __restrict__ img, const unsigned int* __restrict__ Wbf,
        const float* __restrict__ uh, const float* __restrict__ vw,
        const float* __restrict__ vb, float* __restrict__ e_out) {
    __shared__ __align__(16) float uhb[16 * 516];  // uh+biases, de-conflicted stride
    __shared__ float eRed[16];

    const int tid  = threadIdx.x;
    const int lane = tid & 31;
    const int wv   = tid >> 5;          // wave 0..7
    const int m0   = blockIdx.x * 16;   // first flattened row

    // Stage uh for the block's 16 rows (row -> batch via /196)
    #pragma unroll
    for (int j = 0; j < 8; ++j) {
        int idx = tid + 256 * j;        // float4 index 0..2047
        int r   = idx >> 7;
        int c4  = idx & 127;
        int brow = (m0 + r) / N_;
        float4 src = *(const float4*)(uh + brow * H_ + c4 * 4);
        *(float4*)(&uhb[r * 516 + c4 * 4]) = src;
    }
    if (tid < 16) eRed[tid] = vb[0];
    __syncthreads();

    v8f acc[4] = {};

    // per-lane A source: row = lane&15, lanes>=16 take K 16..31 (contiguous, matches B pack)
    const float* aRowBase = img + (long)(m0 + (lane & 15)) * E_ + ((lane >> 4) << 4);

    union Bf { int4 q[2]; v16bf v; };

    #pragma unroll 2
    for (int kb = 0; kb < 64; ++kb) {
        // ---- all 4 B fragments up front: one clause of 8 b128, one wait ----
        const unsigned int* wbBase = Wbf + (((long)kb * 32 + wv * 4) * 32 + lane) * 8;
        Bf b0, b1, b2, b3;
        b0.q[0] = *(const int4*)(wbBase);
        b0.q[1] = *(const int4*)(wbBase + 4);
        b1.q[0] = *(const int4*)(wbBase + 256);
        b1.q[1] = *(const int4*)(wbBase + 260);
        b2.q[0] = *(const int4*)(wbBase + 512);
        b2.q[1] = *(const int4*)(wbBase + 516);
        b3.q[0] = *(const int4*)(wbBase + 768);
        b3.q[1] = *(const int4*)(wbBase + 772);
        if (kb + 1 < 64)   // pull B 4 col-tiles (1024 dwords) ahead from L2 toward the WGP
            __builtin_prefetch(wbBase + 1024, 0, 3);

        // ---- A fragment: 16 fp32 -> 8 packed bf16 dwords (VALU overlaps B) --
        const float* ar = aRowBase + kb * 32;
        float4 a0 = *(const float4*)(ar);
        float4 a1 = *(const float4*)(ar + 4);
        float4 a2 = *(const float4*)(ar + 8);
        float4 a3 = *(const float4*)(ar + 12);
        union { unsigned int u[8]; v16bf v; } af;
        af.u[0] = pack_bf16(a0.x, a0.y);
        af.u[1] = pack_bf16(a0.z, a0.w);
        af.u[2] = pack_bf16(a1.x, a1.y);
        af.u[3] = pack_bf16(a1.z, a1.w);
        af.u[4] = pack_bf16(a2.x, a2.y);
        af.u[5] = pack_bf16(a2.z, a2.w);
        af.u[6] = pack_bf16(a3.x, a3.y);
        af.u[7] = pack_bf16(a3.z, a3.w);

        // ---- 4 back-to-back WMMAs, accumulators updated in place ----
        acc[0] = __builtin_amdgcn_wmma_f32_16x16x32_bf16(
                     false, af.v, false, b0.v, (short)0, acc[0], false, false);
        acc[1] = __builtin_amdgcn_wmma_f32_16x16x32_bf16(
                     false, af.v, false, b1.v, (short)0, acc[1], false, false);
        acc[2] = __builtin_amdgcn_wmma_f32_16x16x32_bf16(
                     false, af.v, false, b2.v, (short)0, acc[2], false, false);
        acc[3] = __builtin_amdgcn_wmma_f32_16x16x32_bf16(
                     false, af.v, false, b3.v, (short)0, acc[3], false, false);

        // schedule shape per stage: loads first, then pack VALU, then 4 WMMAs
        __builtin_amdgcn_sched_group_barrier(0x020, 12, 0);  // VMEM reads
        __builtin_amdgcn_sched_group_barrier(0x002, 24, 0);  // pack VALU
        __builtin_amdgcn_sched_group_barrier(0x008,  4, 0);  // WMMA
    }

    // ---- epilogue: relu(acc + uhb) · v_w, reduced to per-row logits ----
    // C/D layout: VGPR i, lanes 0-15 -> (M=i, N=lane); lanes 16-31 -> (M=8+i).
    float part[8];
    #pragma unroll
    for (int i = 0; i < 8; ++i) part[i] = 0.0f;
    const int mhalf = (lane >> 4) * 8;
    #pragma unroll
    for (int t = 0; t < 4; ++t) {
        int colL  = wv * 64 + t * 16 + (lane & 15);   // 0..511
        float vwv = vw[colL];
        #pragma unroll
        for (int i = 0; i < 8; ++i) {
            float val = acc[t][i] + uhb[(mhalf + i) * 516 + colL];
            val = fmaxf(val, 0.0f);
            part[i] = fmaf(val, vwv, part[i]);
        }
    }
    // reduce over the 16 lanes that share each row (xor masks stay in-half)
    #pragma unroll
    for (int off = 8; off >= 1; off >>= 1) {
        #pragma unroll
        for (int i = 0; i < 8; ++i)
            part[i] += __shfl_xor(part[i], off, 32);
    }
    if ((lane & 15) == 0) {
        #pragma unroll
        for (int i = 0; i < 8; ++i) atomicAdd(&eRed[mhalf + i], part[i]);
    }
    __syncthreads();
    if (tid < 16) e_out[m0 + tid] = eRed[tid];
}

// ---------------------------------------------------------------------------
// Phase 3: softmax over N=196 per batch row
// ---------------------------------------------------------------------------
__global__ __launch_bounds__(256) void softmax_kernel(
        const float* __restrict__ e, float* __restrict__ alpha) {
    __shared__ float red[256];
    int b = blockIdx.x, t = threadIdx.x;
    float x = (t < N_) ? e[b * N_ + t] : -3.4e38f;
    red[t] = x;
    __syncthreads();
    for (int s = 128; s >= 1; s >>= 1) {
        if (t < s) red[t] = fmaxf(red[t], red[t + s]);
        __syncthreads();
    }
    float mx = red[0];
    __syncthreads();
    float ex = (t < N_) ? __expf(x - mx) : 0.0f;
    red[t] = ex;
    __syncthreads();
    for (int s = 128; s >= 1; s >>= 1) {
        if (t < s) red[t] += red[t + s];
        __syncthreads();
    }
    float inv = 1.0f / red[0];
    if (t < N_) alpha[b * N_ + t] = ex * inv;
}

// ---------------------------------------------------------------------------
// Phase 4: context[b,e] = sum_n img[b,n,e] * alpha[b,n]  (bandwidth bound)
// ---------------------------------------------------------------------------
__global__ __launch_bounds__(256) void context_kernel(
        const float* __restrict__ img, const float* __restrict__ alpha,
        float* __restrict__ ctx) {
    __shared__ float al[N_];
    int b  = blockIdx.y;
    int e0 = blockIdx.x * 1024 + threadIdx.x * 4;
    if (threadIdx.x < N_) al[threadIdx.x] = alpha[b * N_ + threadIdx.x];
    __syncthreads();
    float4 acc = {0.f, 0.f, 0.f, 0.f};
    const float* base = img + (long)b * N_ * E_ + e0;
    for (int n = 0; n < N_; ++n) {
        float4 x = *(const float4*)(base + (long)n * E_);
        float a = al[n];
        acc.x = fmaf(x.x, a, acc.x);
        acc.y = fmaf(x.y, a, acc.y);
        acc.z = fmaf(x.z, a, acc.z);
        acc.w = fmaf(x.w, a, acc.w);
    }
    *(float4*)(ctx + (long)b * E_ + e0) = acc;
}

extern "C" void kernel_launch(void* const* d_in, const int* in_sizes, int n_in,
                              void* d_out, int out_size, void* d_ws, size_t ws_size,
                              hipStream_t stream) {
    const float* img = (const float*)d_in[0];   // B*N*E
    const float* hid = (const float*)d_in[1];   // B*H
    const float* Uw  = (const float*)d_in[2];   // H*H
    const float* Ub  = (const float*)d_in[3];   // H
    const float* Ww  = (const float*)d_in[4];   // E*H
    const float* Wb  = (const float*)d_in[5];   // H
    const float* vw  = (const float*)d_in[6];   // H
    const float* vb  = (const float*)d_in[7];   // 1

    float* ctx   = (float*)d_out;                       // B*E
    float* alpha = (float*)d_out + (size_t)B_ * E_;     // B*N

    char* ws = (char*)d_ws;
    float*        uh  = (float*)ws;                     // B*H      = 512 KB
    float*        e   = (float*)(ws + 524288);          // B*N      = 196 KB
    unsigned int* Wbf = (unsigned int*)(ws + 724992);   // E*H bf16 = 2 MB

    pack_w_kernel<<<2048, 256, 0, stream>>>(Ww, Wbf);
    uh_kernel<<<dim3(H_ / 256, B_), 256, 0, stream>>>(hid, Uw, Ub, Wb, uh);
    attn_gemm_kernel<<<(B_ * N_) / 16, 256, 0, stream>>>(img, Wbf, uh, vw, vb, e);
    softmax_kernel<<<B_, 256, 0, stream>>>(e, alpha);
    context_kernel<<<dim3(E_ / 1024, B_), 256, 0, stream>>>(img, alpha, ctx);
}
